// SelfAttention_45037027066543
// MI455X (gfx1250) — matrix operand order repaired and checked by reading, not compiled, not run on hardware
//
#include <hip/hip_runtime.h>
#include <hip/hip_bf16.h>

typedef __attribute__((ext_vector_type(16))) _Float16 v16h;
typedef __attribute__((ext_vector_type(8)))  _Float16 v8h;
typedef __attribute__((ext_vector_type(8)))  float    v8f;
typedef __attribute__((ext_vector_type(4)))  unsigned v4u;
typedef __attribute__((ext_vector_type(8)))  int      v8i;
typedef __attribute__((ext_vector_type(4)))  int      v4i;

#define SEQLEN_C 2048
#define KDIM     1024
#define HEADS_C  16
#define HDIM     64
#define BATCH_C  4
#define ROWS_C   (BATCH_C * SEQLEN_C)   // 8192

// ---------------------------------------------------------------------------
// WMMA helper: D = A(16x32 f16) * B(32x16 f16) + C(16x16 f32)
// ---------------------------------------------------------------------------
static __device__ __forceinline__ v8f wmma_f16f32(v16h a, v16h b, v8f c) {
  return __builtin_amdgcn_wmma_f32_16x16x32_f16(
      /*neg_a=*/false, a, /*neg_b=*/false, b,
      /*c_mod=*/(short)0, c, /*reuse_a=*/false, /*reuse_b=*/false);
}

// ---------------------------------------------------------------------------
// DPP16 lane reductions across 16-lane rows (no LDS traffic, unlike bpermute)
// 0xB1 = quad_perm[1,0,3,2] (xor1), 0x4E = quad_perm[2,3,0,1] (xor2),
// 0x141 = row_half_mirror (crosses 4-groups), 0x140 = row_mirror (crosses 8)
// ---------------------------------------------------------------------------
template <int CTRL>
static __device__ __forceinline__ float dppf(float x) {
  return __int_as_float(__builtin_amdgcn_update_dpp(
      __float_as_int(x), __float_as_int(x), CTRL, 0xF, 0xF, false));
}
static __device__ __forceinline__ float rowmax16(float x) {
  x = fmaxf(x, dppf<0xB1>(x));
  x = fmaxf(x, dppf<0x4E>(x));
  x = fmaxf(x, dppf<0x141>(x));
  x = fmaxf(x, dppf<0x140>(x));
  return x;
}
static __device__ __forceinline__ float rowsum16(float x) {
  x += dppf<0xB1>(x);
  x += dppf<0x4E>(x);
  x += dppf<0x141>(x);
  x += dppf<0x140>(x);
  return x;
}

// ---------------------------------------------------------------------------
// CDNA5 async copy: global -> LDS, ASYNCcnt-tracked (bypasses VGPRs)
// ---------------------------------------------------------------------------
static __device__ __forceinline__ void async_load_b128(unsigned lds_addr,
                                                       const void* gptr) {
  unsigned long long ga = (unsigned long long)(uintptr_t)gptr;
  asm volatile("global_load_async_to_lds_b128 %0, %1, off"
               :: "v"(lds_addr), "v"(ga) : "memory");
}
static __device__ __forceinline__ void wait_async0() {
  asm volatile("s_wait_asynccnt 0" ::: "memory");
}
static __device__ __forceinline__ unsigned lds_off(const void* p) {
  return (unsigned)(uintptr_t)p;   // flat addressing: addr[31:0] = LDS offset
}

// A-matrix fragment (16x32, row-major source [row][k]):
// lane L: m = L&15, kh = L>>4; elems 0..7 -> k = kh*8+0..7, elems 8..15 -> 16+kh*8+0..7
static __device__ __forceinline__ v16h load_frag_a(const _Float16* base, int stride,
                                                   int row, int kbase, int lane) {
  const int kh = (lane >> 4) & 1;
  const _Float16* p = base + (size_t)row * stride + kbase;
  v8h lo = *(const v8h*)(p + kh * 8);
  v8h hi = *(const v8h*)(p + 16 + kh * 8);
  v16h r;
#pragma unroll
  for (int i = 0; i < 8; ++i) { r[i] = lo[i]; r[i + 8] = hi[i]; }
  return r;
}

// B-matrix fragment (32x16), source stored as [n][k] row-major:
// lane L: n = L&15, kh = L>>4; elems 0..7 -> k = kh*8+0..7, elems 8..15 -> 16+kh*8+0..7
static __device__ __forceinline__ v16h load_frag_b(const _Float16* base, int stride,
                                                   int row, int kbase, int lane) {
  const int kh = (lane >> 4) & 1;
  const _Float16* p = base + (size_t)row * stride + kbase + kh * 8;
  v8h lo = *(const v8h*)(p);
  v8h hi = *(const v8h*)(p + 16);
  v16h r;
#pragma unroll
  for (int i = 0; i < 8; ++i) { r[i] = lo[i]; r[i + 8] = hi[i]; }
  return r;
}

// ---------------------------------------------------------------------------
// f32 -> f16 conversion
// ---------------------------------------------------------------------------
__global__ void cvt_f32_to_f16(const float* __restrict__ src,
                               _Float16* __restrict__ dst, int n) {
  int i = blockIdx.x * blockDim.x + threadIdx.x;
  if (i < n) dst[i] = (_Float16)src[i];
}

// ---------------------------------------------------------------------------
// GEMM: C[M,N] = A[M,Kd] * Bw[N,Kd]^T (+ bias). BM=128, BN=128, BK=32.
// 256 threads = 8 waves in a 4(m) x 2(n) grid; each wave computes 32x64.
// LDS tiles filled with global_load_async_to_lds_b128.
// ---------------------------------------------------------------------------
template <typename OutT>
__global__ __launch_bounds__(256) void gemm_wmma(const _Float16* __restrict__ A,
                                                 const _Float16* __restrict__ Bw,
                                                 const float* __restrict__ bias,
                                                 OutT* __restrict__ C,
                                                 int M, int N, int Kd) {
  __shared__ _Float16 As[128 * 32];
  __shared__ _Float16 Bs[128 * 32];
  const int tid = threadIdx.x;
  const int wave = tid >> 5, lane = tid & 31;
  const int waveM = wave & 3, waveN = wave >> 2;
  const int rowTile = blockIdx.y * 128, colTile = blockIdx.x * 128;
  const int nl = lane & 15, halfSel = lane >> 4;

  v8f vzero = {};
  v8f acc[2][4];
#pragma unroll
  for (int a = 0; a < 2; ++a)
#pragma unroll
    for (int b = 0; b < 4; ++b) acc[a][b] = vzero;

  for (int kb = 0; kb < Kd; kb += 32) {
    // Async-stage 128x32 tiles of A and Bw into LDS (16B per thread per tile x2).
#pragma unroll
    for (int i = tid; i < 512; i += 256) {
      int r = i >> 2, c = (i & 3) * 8;
      async_load_b128(lds_off(&As[r * 32 + c]),
                      &A[(size_t)(rowTile + r) * Kd + kb + c]);
      async_load_b128(lds_off(&Bs[r * 32 + c]),
                      &Bw[(size_t)(colTile + r) * Kd + kb + c]);
    }
    wait_async0();
    __syncthreads();

    v16h af[2];
#pragma unroll
    for (int tm = 0; tm < 2; ++tm)
      af[tm] = load_frag_a(As, 32, waveM * 32 + tm * 16 + nl, 0, lane);
    v16h bf[4];
#pragma unroll
    for (int tn = 0; tn < 4; ++tn)
      bf[tn] = load_frag_b(Bs, 32, waveN * 64 + tn * 16 + nl, 0, lane);
#pragma unroll
    for (int tm = 0; tm < 2; ++tm)
#pragma unroll
      for (int tn = 0; tn < 4; ++tn)
        acc[tm][tn] = wmma_f16f32(af[tm], bf[tn], acc[tm][tn]);
    __syncthreads();
  }

  // C/D layout: m = v + 8*(lane>>4), n = lane&15
#pragma unroll
  for (int tm = 0; tm < 2; ++tm)
#pragma unroll
    for (int tn = 0; tn < 4; ++tn)
#pragma unroll
      for (int v = 0; v < 8; ++v) {
        int gm = rowTile + waveM * 32 + tm * 16 + v + 8 * halfSel;
        int gn = colTile + waveN * 64 + tn * 16 + nl;
        float val = acc[tm][tn][v];
        if (bias) val += bias[gn];
        C[(size_t)gm * N + gn] = (OutT)val;
      }
}

// ---------------------------------------------------------------------------
// RoPE on Q and K in place (f16 storage, f32 math). One thread per pair.
// ---------------------------------------------------------------------------
__global__ void rope_f16(_Float16* __restrict__ Q, _Float16* __restrict__ Kt, int total) {
  int i = blockIdx.x * blockDim.x + threadIdx.x;
  if (i >= total) return;
  const int half = KDIM / 2;                  // 512
  int j = i & (half - 1);
  int row = i >> 9;
  int t = row & (SEQLEN_C - 1);
  float freq = __expf(-(float)j * (9.210340371976184f / (float)half));
  float ang = (float)t * freq;
  float sn, cs;
  __sincosf(ang, &sn, &cs);
  size_t off = (size_t)row * KDIM + 2 * j;
  float qe = (float)Q[off], qo = (float)Q[off + 1];
  Q[off]     = (_Float16)(qe * cs - qo * sn);
  Q[off + 1] = (_Float16)(qe * sn + qo * cs);
  float ke = (float)Kt[off], ko = (float)Kt[off + 1];
  Kt[off]     = (_Float16)(ke * cs - ko * sn);
  Kt[off + 1] = (_Float16)(ke * sn + ko * cs);
}

// ---------------------------------------------------------------------------
// Flash attention: grid (SEQ/64, B*H), 128 threads = 4 waves.
// K chunk staged by the Tensor Data Mover (wave 0 issues one 2D-tile D#),
// V chunk staged transposed by all threads, P relayout via per-wave LDS.
// ---------------------------------------------------------------------------
__global__ __launch_bounds__(128) void attn_flash(const _Float16* __restrict__ Q,
                                                  const _Float16* __restrict__ Kc,
                                                  const _Float16* __restrict__ V,
                                                  _Float16* __restrict__ O) {
  __shared__ _Float16 Ks[64 * 64];         // K chunk, [key][d]  (TDM target)
  __shared__ _Float16 VTs[64 * 64];        // V chunk transposed, [d][key]
  __shared__ _Float16 Ps[4][16 * 64];      // per-wave P staging

  const int tid = threadIdx.x;
  const int wave = tid >> 5, lane = tid & 31;
  const int nl = lane & 15, halfSel = lane >> 4;
  const int qt = blockIdx.x;               // query tile, 0..31
  const int bh = blockIdx.y;               // 0..63
  const int b = bh >> 4, h = bh & 15;
  const int hc = h * HDIM;
  const int browBase = b * SEQLEN_C;

  // Q fragments (A layout), held in registers across all KV chunks
  v16h qf[2];
  {
    int qrow = browBase + qt * 64 + wave * 16 + nl;
#pragma unroll
    for (int tk = 0; tk < 2; ++tk)
      qf[tk] = load_frag_a(Q, KDIM, qrow, hc + tk * 32, lane);
  }

  v8f vzero = {};
  v8f oacc[4];
#pragma unroll
  for (int td = 0; td < 4; ++td) oacc[td] = vzero;
  float run_m[8], run_l[8];
#pragma unroll
  for (int v = 0; v < 8; ++v) { run_m[v] = -1e30f; run_l[v] = 0.0f; }

  _Float16* pw = Ps[wave];
  const float scale = 0.125f;              // 1/sqrt(64)
  const unsigned ksLds = lds_off(&Ks[0]);

  for (int c = 0; c < SEQLEN_C / 64; ++c) {
    // --- K chunk via Tensor Data Mover: 64x64 f16 tile, row stride KDIM ---
    if (wave == 0) {
      unsigned long long ga = (unsigned long long)(uintptr_t)
          (Kc + (size_t)(browBase + c * 64) * KDIM + hc);
      v4u g0;
      g0[0] = 1u;                                   // count=1 (valid user D#)
      g0[1] = ksLds;                                // lds_addr
      g0[2] = (unsigned)(ga & 0xFFFFFFFFu);         // global_addr[31:0]
      g0[3] = (unsigned)((ga >> 32) & 0x01FFFFFFu)  // global_addr[56:32]
              | 0x80000000u;                        // type=2 ("image")
      v8i g1;
      g1[0] = (int)(1u << 16);                      // data_size=1 -> 2 bytes
      g1[1] = (int)((KDIM & 0xFFFFu) << 16);        // tensor_dim0[15:0]=1024
      g1[2] = (int)((KDIM >> 16) | ((ROWS_C & 0xFFFFu) << 16)); // td0 hi | td1 lo
      g1[3] = (int)((ROWS_C >> 16) | (64u << 16));  // td1 hi | tile_dim0=64
      g1[4] = (int)(64u);                           // tile_dim1=64, tile_dim2=0
      g1[5] = (int)KDIM;                            // tensor_dim0_stride=1024
      g1[6] = 0;
      g1[7] = 0;
      v4i g2 = {0, 0, 0, 0};
      v4i g3 = {0, 0, 0, 0};
      v8i g4 = {0, 0, 0, 0, 0, 0, 0, 0};            // 6-arg toolchain: extra group
      __builtin_amdgcn_tensor_load_to_lds(g0, g1, g2, g3, g4, 0);
      __builtin_amdgcn_s_wait_tensorcnt(0);
    }
    // --- V chunk staged transposed (all 128 threads) ---
#pragma unroll
    for (int i = tid; i < 512; i += 128) {
      int r = i >> 3, cc = (i & 7) * 8;
      size_t g = (size_t)(browBase + c * 64 + r) * KDIM + hc + cc;
      v8h vv = *(const v8h*)&V[g];
#pragma unroll
      for (int jj = 0; jj < 8; ++jj) VTs[(cc + jj) * 64 + r] = vv[jj];
    }
    __syncthreads();

    // S = Q * K^T, one 16x64 strip per wave
    v8f s[4];
#pragma unroll
    for (int tn = 0; tn < 4; ++tn) {
      v8f z = vzero;
#pragma unroll
      for (int tk = 0; tk < 2; ++tk) {
        v16h kf = load_frag_b(Ks, 64, tn * 16 + nl, tk * 32, lane);
        z = wmma_f16f32(qf[tk], kf, z);
      }
      s[tn] = z;
    }

#pragma unroll
    for (int tn = 0; tn < 4; ++tn)
#pragma unroll
      for (int v = 0; v < 8; ++v) s[tn][v] *= scale;

    // Online softmax (DPP16 row reductions; row m = v + 8*halfSel over 16 lanes)
#pragma unroll
    for (int v = 0; v < 8; ++v) {
      float m0 = fmaxf(fmaxf(s[0][v], s[1][v]), fmaxf(s[2][v], s[3][v]));
      m0 = rowmax16(m0);
      float nm = fmaxf(run_m[v], m0);
      float corr = __expf(run_m[v] - nm);
      run_m[v] = nm;
      float rs = 0.0f;
#pragma unroll
      for (int tn = 0; tn < 4; ++tn) {
        float p = __expf(s[tn][v] - nm);
        s[tn][v] = p;
        rs += p;
      }
      rs = rowsum16(rs);
      run_l[v] = run_l[v] * corr + rs;
#pragma unroll
      for (int td = 0; td < 4; ++td) oacc[td][v] *= corr;
    }

    // Re-layout P (C layout -> A layout) through per-wave LDS.
#pragma unroll
    for (int tn = 0; tn < 4; ++tn)
#pragma unroll
      for (int v = 0; v < 8; ++v)
        pw[(v + 8 * halfSel) * 64 + tn * 16 + nl] = (_Float16)s[tn][v];
    __builtin_amdgcn_wave_barrier();

    // O += P * V   (V^T as B so the fragment gather is contiguous)
#pragma unroll
    for (int tk = 0; tk < 2; ++tk) {
      v16h pf = load_frag_a(pw, 64, nl, tk * 32, lane);
#pragma unroll
      for (int td = 0; td < 4; ++td) {
        v16h vf = load_frag_b(VTs, 64, td * 16 + nl, tk * 32, lane);
        oacc[td] = wmma_f16f32(pf, vf, oacc[td]);
      }
    }
    __syncthreads();
  }

  // Normalize and store O (f16)
#pragma unroll
  for (int td = 0; td < 4; ++td)
#pragma unroll
    for (int v = 0; v < 8; ++v) {
      int gr = browBase + qt * 64 + wave * 16 + v + 8 * halfSel;
      int gc = hc + td * 16 + nl;
      O[(size_t)gr * KDIM + gc] = (_Float16)(oacc[td][v] / run_l[v]);
    }
}

// ---------------------------------------------------------------------------
// Launch
// ---------------------------------------------------------------------------
extern "C" void kernel_launch(void* const* d_in, const int* in_sizes, int n_in,
                              void* d_out, int out_size, void* d_ws, size_t ws_size,
                              hipStream_t stream) {
  (void)in_sizes; (void)n_in; (void)out_size; (void)ws_size;
  const float* x  = (const float*)d_in[0];
  const float* Wq = (const float*)d_in[1];
  const float* Wk = (const float*)d_in[2];
  const float* Wv = (const float*)d_in[3];
  const float* Wu = (const float*)d_in[4];
  const float* bu = (const float*)d_in[5];
  float* out = (float*)d_out;

  char* ws = (char*)d_ws;
  const size_t MB = (size_t)1 << 20;
  _Float16* xh  = (_Float16*)(ws + 0 * MB);    // 16 MB  (8192x1024 f16)
  _Float16* Wqh = (_Float16*)(ws + 16 * MB);   // 2 MB
  _Float16* Wkh = (_Float16*)(ws + 18 * MB);   // 2 MB
  _Float16* Wvh = (_Float16*)(ws + 20 * MB);   // 2 MB
  _Float16* Wuh = (_Float16*)(ws + 22 * MB);   // 2 MB
  _Float16* Qh  = (_Float16*)(ws + 24 * MB);   // 16 MB
  _Float16* Kh  = (_Float16*)(ws + 40 * MB);   // 16 MB
  _Float16* Vh  = (_Float16*)(ws + 56 * MB);   // 16 MB
  _Float16* Oh  = (_Float16*)(ws + 72 * MB);   // 16 MB

  const int nX = ROWS_C * KDIM;                // 8388608
  const int nW = KDIM * KDIM;                  // 1048576
  cvt_f32_to_f16<<<(nX + 255) / 256, 256, 0, stream>>>(x, xh, nX);
  cvt_f32_to_f16<<<(nW + 255) / 256, 256, 0, stream>>>(Wq, Wqh, nW);
  cvt_f32_to_f16<<<(nW + 255) / 256, 256, 0, stream>>>(Wk, Wkh, nW);
  cvt_f32_to_f16<<<(nW + 255) / 256, 256, 0, stream>>>(Wv, Wvh, nW);
  cvt_f32_to_f16<<<(nW + 255) / 256, 256, 0, stream>>>(Wu, Wuh, nW);

  dim3 gg(KDIM / 128, ROWS_C / 128);           // (8, 64)
  gemm_wmma<_Float16><<<gg, 256, 0, stream>>>(xh, Wqh, nullptr, Qh, ROWS_C, KDIM, KDIM);
  gemm_wmma<_Float16><<<gg, 256, 0, stream>>>(xh, Wkh, nullptr, Kh, ROWS_C, KDIM, KDIM);
  gemm_wmma<_Float16><<<gg, 256, 0, stream>>>(xh, Wvh, nullptr, Vh, ROWS_C, KDIM, KDIM);

  const int nR = ROWS_C * (KDIM / 2);          // 4194304 pairs
  rope_f16<<<(nR + 255) / 256, 256, 0, stream>>>(Qh, Kh, nR);

  dim3 ga(SEQLEN_C / 64, BATCH_C * HEADS_C);   // (32, 64)
  attn_flash<<<ga, 128, 0, stream>>>(Qh, Kh, Vh, Oh);

  gemm_wmma<float><<<gg, 256, 0, stream>>>(Oh, Wuh, bu, out, ROWS_C, KDIM, KDIM);
}
